// TemporalAttention_78460462563674
// MI455X (gfx1250) — matrix operand order repaired and checked by reading, not compiled, not run on hardware
//
#include <hip/hip_runtime.h>

// ---- MI455X / gfx1250 fused temporal-attention, TDM edition ----
// out[d,b] = sum_s softmax_s(u * (enc[b,s,:]·We_w)) * enc[b,s,d]
// Single pass over enc_out (256 MiB @ 23.3 TB/s ~= 11 us floor).
// Per (b, 32-row s-chunk) block:
//   TENSOR_LOAD_TO_LDS (async DMA, descriptor-padded rows) -> f32 LDS tile
//   WMMA f32_16x16x32_f16 row-dots -> stable-softmax partial (M, Z, acc[256])
// Merge kernel rescales 128 chunk-partials per batch.

typedef __attribute__((ext_vector_type(16))) _Float16 v16h;
typedef __attribute__((ext_vector_type(8)))  float    v8f;
typedef __attribute__((ext_vector_type(2)))  float    f2;
typedef __attribute__((ext_vector_type(4)))  unsigned v4u;
typedef __attribute__((ext_vector_type(8)))  int      v8i;
typedef __attribute__((ext_vector_type(4)))  int      v4i;

#define B_      64
#define S_      4096
#define D_      256
#define ROWS    32
#define CHUNKS  (S_ / ROWS)        // 128
#define TSTRIDE 268                // f32 tile row stride in dwords (256 + 12 pad via D#)
#define TILE_BYTES (ROWS * TSTRIDE * 4)   // 34304
#define REC     (D_ + 2)           // partial record: acc[256], M, Z

// LDS blob layout (tile MUST sit at LDS offset 0: the D# encodes lds_addr=0)
#define OFF_WV   TILE_BYTES              // _Float16[256], 32B-aligned (34304 % 32 == 0)
#define OFF_SC   (OFF_WV + 512)          // float[32]
#define OFF_EV   (OFF_SC + 128)          // float[32]
#define OFF_RED  (OFF_EV + 128)          // float[16]
#define LDS_TOTAL (OFF_RED + 64)

__global__ __launch_bounds__(128) void ta_fused(
    const float* __restrict__ enc,   // [B,S,D]
    const float* __restrict__ Ww,    // [D]
    const float* __restrict__ ut,    // [1]
    float* __restrict__ P)           // [B][CHUNKS][REC]
{
    __shared__ __align__(128) unsigned char smem[LDS_TOTAL];
    float*    tile = (float*)smem;
    _Float16* wv   = (_Float16*)(smem + OFF_WV);
    float*    sc   = (float*)(smem + OFF_SC);
    float*    ev   = (float*)(smem + OFF_EV);
    float*    red  = (float*)(smem + OFF_RED);

    const int t     = threadIdx.x;   // 0..127
    const int chunk = blockIdx.x;    // 0..127
    const int b     = blockIdx.y;    // 0..63
    const float u   = ut[0];

    // ---- Phase 1: TDM async DMA of 32 rows x 256 f32 into LDS ----
    // D# pad: every 256 dwords (one row) insert 12 dwords -> LDS row stride 268 dw.
    // 12*r mod 64 is injective for r in [0,16) -> bank-spread rows for WMMA gathers.
    if (t < 32) {   // one wave issues the tensor op (EXEC ignored by TDM)
        unsigned long long ga = (unsigned long long)
            (enc + ((size_t)b * S_ + (size_t)chunk * ROWS) * D_);
        v4u g0 = { 1u,                                   // count=1, user mode
                   0u,                                   // lds_addr = 0
                   (unsigned)(ga & 0xffffffffu),         // global_addr[31:0]
                   (unsigned)((ga >> 32) & 0x01ffffffu)  // global_addr[56:32]
                     | (2u << 30) };                     // type = 2 ("image")
        v8i g1 = { (int)((2u << 16)                      // data_size = 4B
                     | (1u << 20)                        // pad_enable
                     | (7u << 22)                        // pad_interval: 256 dwords
                     | (11u << 25)),                     // pad_amount: 12 dwords
                   (int)(256u << 16),                    // tensor_dim0 = 256 (lo16)
                   (int)(32u  << 16),                    // dim0 hi=0 | tensor_dim1 = 32 (lo16)
                   (int)(256u << 16),                    // dim1 hi=0 | tile_dim0 = 256
                   (int)(32u),                           // tile_dim1 = 32, tile_dim2 = 0
                   (int)(256u),                          // tensor_dim0_stride = 256
                   0, 0 };
        v4i gz4 = {0, 0, 0, 0};                          // 2-D: groups 2/3 unused
        v8i gz8 = {0, 0, 0, 0, 0, 0, 0, 0};              // clang-23 6-arg form
        __builtin_amdgcn_tensor_load_to_lds(g0, g1, gz4, gz4, gz8, 0);
        __builtin_amdgcn_s_wait_tensorcnt(0);
    }
    wv[t]       = (_Float16)Ww[t];
    wv[t + 128] = (_Float16)Ww[t + 128];
    __syncthreads();

    // ---- Phase 2: WMMA row dots. Waves 0-1, 16 rows each, K=256 in 8 steps. ----
    // A (16x32 f16) element e=2j+h of lane (m=lane&15, hi=lane>>4):
    //   K = ks*32 + (hi?16:0) + 2*j + h   (contiguous, matches B)
    // B (32x16 f16): Ww[k] replicated across all 16 columns -> every D column = dot.
    const int lane = t & 31;
    const int m    = lane & 15;
    const int hi   = lane >> 4;
    if (t < 64) {
        const int r0 = (t >> 5) * 16;
        v8f c = {0.f, 0.f, 0.f, 0.f, 0.f, 0.f, 0.f, 0.f};
        #pragma unroll
        for (int ks = 0; ks < 8; ++ks) {
            v16h a;
            #pragma unroll
            for (int j = 0; j < 8; ++j) {
                int koff = ks * 32 + (hi ? 16 : 0) + 2 * j;
                f2 p = *(const f2*)(tile + (r0 + m) * TSTRIDE + koff);
                a[2 * j]     = (_Float16)p.x;
                a[2 * j + 1] = (_Float16)p.y;
            }
            v16h bf = *(const v16h*)(wv + ks * 32 + hi * 16);
            c = __builtin_amdgcn_wmma_f32_16x16x32_f16(
                    /*neg_a=*/false, a, /*neg_b=*/false, bf,
                    /*c_mod=*/(short)0, c, /*reuse_a=*/false, /*reuse_b=*/false);
        }
        // C/D layout: lanes 0-15 hold M=0..7 in c[0..7]; lanes 16-31 hold M=8..15.
        if (lane == 0) {
            #pragma unroll
            for (int mm = 0; mm < 8; ++mm) sc[r0 + mm] = c[mm] * u;
        }
        if (lane == 16) {
            #pragma unroll
            for (int mm = 0; mm < 8; ++mm) sc[r0 + 8 + mm] = c[mm] * u;
        }
    }
    __syncthreads();

    // ---- Phase 3: chunk max over 32 scores ----
    if (t < 16) red[t] = fmaxf(sc[t], sc[t + 16]);
    __syncthreads();
    if (t < 8)  red[t] = fmaxf(red[t], red[t + 8]);
    __syncthreads();
    if (t < 4)  red[t] = fmaxf(red[t], red[t + 4]);
    __syncthreads();
    if (t < 2)  red[t] = fmaxf(red[t], red[t + 2]);
    __syncthreads();
    if (t < 1)  red[0] = fmaxf(red[0], red[1]);
    __syncthreads();
    const float M = red[0];
    __syncthreads();

    // ---- Phase 4: exp + chunk Z ----
    if (t < 32) ev[t] = __expf(sc[t] - M);
    __syncthreads();
    if (t < 16) red[t] = ev[t] + ev[t + 16];
    __syncthreads();
    if (t < 8)  red[t] = red[t] + red[t + 8];
    __syncthreads();
    if (t < 4)  red[t] = red[t] + red[t + 4];
    __syncthreads();
    if (t < 2)  red[t] = red[t] + red[t + 2];
    __syncthreads();
    if (t < 1)  red[0] = red[0] + red[1];
    __syncthreads();
    const float Z = red[0];

    // ---- Phase 5: weighted accumulate from f32 LDS tile, 2 cols/thread ----
    float acc0 = 0.f, acc1 = 0.f;
    #pragma unroll 4
    for (int s = 0; s < ROWS; ++s) {
        float e = ev[s];
        acc0 += e * tile[s * TSTRIDE + t];
        acc1 += e * tile[s * TSTRIDE + 128 + t];
    }

    size_t rec = ((size_t)b * CHUNKS + chunk) * REC;
    P[rec + t]       = acc0;
    P[rec + 128 + t] = acc1;
    if (t == 0) { P[rec + D_] = M; P[rec + D_ + 1] = Z; }
}

__global__ __launch_bounds__(256) void ta_reduce(
    const float* __restrict__ P,   // [B][CHUNKS][REC]
    float* __restrict__ out)       // [D,B]
{
    const int b = blockIdx.x;      // 0..63
    const int d = threadIdx.x;     // 0..255

    const float* Pb = P + (size_t)b * CHUNKS * REC;

    float gM = -3.402823466e38f;
    #pragma unroll 8
    for (int c = 0; c < CHUNKS; ++c)
        gM = fmaxf(gM, Pb[(size_t)c * REC + D_]);

    float num = 0.f, den = 0.f;
    #pragma unroll 4
    for (int c = 0; c < CHUNKS; ++c) {
        const float* r = Pb + (size_t)c * REC;
        float w = __expf(r[D_] - gM);
        num += r[d] * w;
        den += r[D_ + 1] * w;
    }
    out[(size_t)d * B_ + b] = num / den;
}

extern "C" void kernel_launch(void* const* d_in, const int* in_sizes, int n_in,
                              void* d_out, int out_size, void* d_ws, size_t ws_size,
                              hipStream_t stream) {
    (void)in_sizes; (void)n_in; (void)out_size; (void)ws_size;
    const float* enc = (const float*)d_in[0];   // [64,4096,256] f32
    const float* Ww  = (const float*)d_in[1];   // [1,256] f32
    // d_in[2] = We_b: constant shift over the softmax axis -> cancels, unused.
    const float* ut  = (const float*)d_in[3];   // [1,1] f32
    float* P   = (float*)d_ws;                  // 64*128*258 floats ~= 8.5 MB
    float* out = (float*)d_out;                 // [256,64] f32

    dim3 grid(CHUNKS, B_);                      // (128 s-chunks, 64 batches)
    ta_fused<<<grid, 128, 0, stream>>>(enc, Ww, ut, P);
    ta_reduce<<<B_, D_, 0, stream>>>(P, out);
}
